// MoEFFN_10746008174600
// MI455X (gfx1250) — compile-verified, matrix operand order checked by reading, not run on hardware
//
#include <hip/hip_runtime.h>
#include <hip/hip_bf16.h>
#include <cstdint>

// ---------------- problem constants ----------------
#define N_EXPERTS 8
#define TOP_K     2
#define D_MODEL   1024
#define D_FF      2048
#define N_TOK     8192              // B*T = 4*2048
#define MAX_ROWS  16512             // padded expert-major rows: 2N + 8*15 rounded up to /16
#define ROW_TILES (MAX_ROWS / 16)   // 1032

// meta[] int layout in workspace:
//  [0..7]   per-expert counts
//  [8..15]  per-expert write cursors
//  [16..24] padded segment offsets (off[0..8])
//  [25]     total padded rows
#define M_CNT 0
#define M_CUR 8
#define M_OFF 16
#define M_TOT 25

typedef __bf16        v16bf __attribute__((ext_vector_type(16)));
typedef float         v8f   __attribute__((ext_vector_type(8)));
typedef unsigned int  u32x4 __attribute__((ext_vector_type(4)));

union Frag16 { v16bf v; u32x4 q[2]; };

__device__ __forceinline__ unsigned short f2bf(float f) {
    // round-to-nearest-even fp32 -> bf16
    unsigned int u = __float_as_uint(f);
    unsigned int r = (u + 0x7FFFu + ((u >> 16) & 1u)) >> 16;
    return (unsigned short)r;
}

// ---------------- 1) router: logits, top-2 softmax, counts ----------------
__global__ __launch_bounds__(256) void k_router(const float* __restrict__ x,
                                                const float* __restrict__ rw,
                                                int* __restrict__ meta,
                                                int* __restrict__ tok_e,
                                                float* __restrict__ tok_w) {
    const int lane = threadIdx.x & 31;
    const int wave = threadIdx.x >> 5;
    const int t    = blockIdx.x * 8 + wave;   // one wave32 per token
    float acc[N_EXPERTS];
#pragma unroll
    for (int e = 0; e < N_EXPERTS; ++e) acc[e] = 0.0f;
    const float* xr = x + (size_t)t * D_MODEL;
    for (int d = lane; d < D_MODEL; d += 32) {
        float xv = xr[d];
#pragma unroll
        for (int e = 0; e < N_EXPERTS; ++e) acc[e] += xv * rw[e * D_MODEL + d];
    }
#pragma unroll
    for (int m = 16; m >= 1; m >>= 1) {
#pragma unroll
        for (int e = 0; e < N_EXPERTS; ++e) acc[e] += __shfl_xor(acc[e], m, 32);
    }
    if (lane == 0) {
        int i0 = 0; float v0 = acc[0];
#pragma unroll
        for (int e = 1; e < N_EXPERTS; ++e) if (acc[e] > v0) { v0 = acc[e]; i0 = e; }
        int i1 = -1; float v1 = -3.4e38f;
#pragma unroll
        for (int e = 0; e < N_EXPERTS; ++e) if (e != i0 && acc[e] > v1) { v1 = acc[e]; i1 = e; }
        float ex = __expf(v1 - v0);            // v1 <= v0, stable
        float w0 = 1.0f / (1.0f + ex);
        float w1 = ex * w0;
        tok_e[2 * t]     = i0;  tok_w[2 * t]     = w0;
        tok_e[2 * t + 1] = i1;  tok_w[2 * t + 1] = w1;
        atomicAdd(&meta[M_CNT + i0], 1);
        atomicAdd(&meta[M_CNT + i1], 1);
    }
}

// ---------------- 2) padded prefix sum over 8 experts ----------------
__global__ void k_offsets(int* __restrict__ meta) {
    meta[M_OFF] = 0;
    for (int e = 0; e < N_EXPERTS; ++e) {
        int padded = (meta[M_CNT + e] + 15) & ~15;
        meta[M_OFF + e + 1] = meta[M_OFF + e] + padded;
        meta[M_CUR + e]     = meta[M_OFF + e];
    }
    meta[M_TOT] = meta[M_OFF + N_EXPERTS];
}

// ---------------- 3) gather tokens expert-major (fp32 -> bf16) ----------------
__global__ __launch_bounds__(64) void k_gather(const float* __restrict__ x,
                                               const int* __restrict__ tok_e,
                                               const float* __restrict__ tok_w,
                                               int* __restrict__ meta,
                                               int* __restrict__ row_token,
                                               float* __restrict__ wrow,
                                               unsigned short* __restrict__ xg,
                                               unsigned short* __restrict__ xb) {
    const int t = blockIdx.x;
    __shared__ int s_pos[2];
    if (threadIdx.x == 0) {
        int e0 = tok_e[2 * t], e1 = tok_e[2 * t + 1];
        int p0 = atomicAdd(&meta[M_CUR + e0], 1);
        int p1 = atomicAdd(&meta[M_CUR + e1], 1);
        s_pos[0] = p0; s_pos[1] = p1;
        row_token[p0] = t; wrow[p0] = tok_w[2 * t];
        row_token[p1] = t; wrow[p1] = tok_w[2 * t + 1];
    }
    __syncthreads();
    const int p0 = s_pos[0], p1 = s_pos[1];
    const float* xr = x + (size_t)t * D_MODEL;
    for (int j = threadIdx.x; j < D_MODEL; j += 64) {
        unsigned short b = f2bf(xr[j]);
        xg[(size_t)p0 * D_MODEL + j] = b;
        xg[(size_t)p1 * D_MODEL + j] = b;
        xb[(size_t)t  * D_MODEL + j] = b;
    }
}

// ---------------- 4) transpose + convert weights: dst[c*rows+r] = src[r*cols+c] ----------------
__global__ __launch_bounds__(256) void k_tcvt(const float* __restrict__ src,
                                              unsigned short* __restrict__ dst,
                                              int rows, int cols) {
    const size_t mat = (size_t)blockIdx.z * rows * cols;
    const size_t i   = (size_t)blockIdx.x * 256 + threadIdx.x;
    if (i >= (size_t)rows * cols) return;
    const int c = (int)(i / rows);
    const int r = (int)(i % rows);
    dst[mat + i] = f2bf(src[mat + (size_t)r * cols + c]);
}

// ---------------- 5) fused gate/up GEMM + SiLU  (h = silu(x@Wg) * (x@Wu)) ----------------
// A: [rows, D_MODEL] bf16 row-major.  Wg/Wu: [e][F][D] bf16 (K-contiguous).  H: [rows, F] bf16.
// Block = 4 waves; each wave computes a 16-row x 64-col tile (4 col sub-tiles reuse one A frag).
// waves_per_eu(2,4) relaxes the RA occupancy target so all 9 fragments of a K-step can hold
// distinct VGPRs -> loads stay in flight across the WMMA chain (graduated s_wait_loadcnt).
__global__ __launch_bounds__(128)
__attribute__((amdgpu_waves_per_eu(2, 4)))
void k_gateup(const unsigned short* __restrict__ A,
              const unsigned short* __restrict__ Wg,
              const unsigned short* __restrict__ Wu,
              unsigned short* __restrict__ H,
              const int* __restrict__ meta,
              int expert_mode, int fixed_rows) {
    const int lane = threadIdx.x & 31;
    const int wave = threadIdx.x >> 5;
    const int rowbase = blockIdx.x * 16;
    const int total = expert_mode ? meta[M_TOT] : fixed_rows;
    if (rowbase >= total) return;                      // uniform: EXEC stays all-1s for WMMA
    int e = 0;
    if (expert_mode) { while (e < N_EXPERTS - 1 && rowbase >= meta[M_OFF + e + 1]) ++e; }
    const size_t wb = (size_t)e * D_MODEL * D_FF;

    const int lr = lane & 15, hi = lane >> 4;
    const int colbase = blockIdx.y * 256 + wave * 64;  // 4 waves x 64 cols = 256 cols / block
    const unsigned short* ap = A  + (size_t)(rowbase + lr) * D_MODEL;
    const unsigned short* gp = Wg + wb + (size_t)(colbase + lr) * D_MODEL;
    const unsigned short* up = Wu + wb + (size_t)(colbase + lr) * D_MODEL;

    v8f accg[4] = {};
    v8f accu[4] = {};
    for (int k0 = 0; k0 < D_MODEL; k0 += 32) {
        const int c0 = k0 + hi * 8, c1 = c0 + 16;
        Frag16 a, bg[4], bu[4];
        // issue all loads for this K-step before any WMMA consumes them
        a.q[0] = *(const u32x4*)(ap + c0);
        a.q[1] = *(const u32x4*)(ap + c1);
#pragma unroll
        for (int j = 0; j < 4; ++j) {
            const size_t joff = (size_t)j * 16 * D_MODEL;
            bg[j].q[0] = *(const u32x4*)(gp + joff + c0);
            bg[j].q[1] = *(const u32x4*)(gp + joff + c1);
            bu[j].q[0] = *(const u32x4*)(up + joff + c0);
            bu[j].q[1] = *(const u32x4*)(up + joff + c1);
        }
#pragma unroll
        for (int j = 0; j < 4; ++j) {
            accg[j] = __builtin_amdgcn_wmma_f32_16x16x32_bf16(false, a.v, false, bg[j].v,
                                                              (short)0, accg[j], false, false);
            accu[j] = __builtin_amdgcn_wmma_f32_16x16x32_bf16(false, a.v, false, bu[j].v,
                                                              (short)0, accu[j], false, false);
        }
    }
#pragma unroll
    for (int j = 0; j < 4; ++j) {
        const int colo = colbase + j * 16 + lr;
#pragma unroll
        for (int i = 0; i < 8; ++i) {
            float g = accg[j][i], u = accu[j][i];
            // SiLU(g)*u with v_rcp_f32 instead of IEEE division (bf16 output anyway)
            float s = __builtin_amdgcn_rcpf(1.0f + __expf(-g));
            float h = g * s * u;
            const int row = rowbase + i + hi * 8;      // C-frag: lanes 0-15 rows 0-7, 16-31 rows 8-15
            H[(size_t)row * D_FF + colo] = f2bf(h);
        }
    }
}

// ---------------- 6) down GEMM + weighted scatter into out ----------------
// H: [rows, F] bf16.  Wd: [e][D][F] bf16 (K-contiguous).  out: [N_TOK, D_MODEL] fp32.
// Block = 4 waves; each wave computes a 16-row x 64-col tile.
__global__ __launch_bounds__(128)
__attribute__((amdgpu_waves_per_eu(2, 4)))
void k_down(const unsigned short* __restrict__ H,
            const unsigned short* __restrict__ Wd,
            float* __restrict__ out,
            const int* __restrict__ meta,
            const int* __restrict__ row_token,
            const float* __restrict__ wrow,
            int expert_mode, int fixed_rows) {
    const int lane = threadIdx.x & 31;
    const int wave = threadIdx.x >> 5;
    const int rowbase = blockIdx.x * 16;
    const int total = expert_mode ? meta[M_TOT] : fixed_rows;
    if (rowbase >= total) return;
    int e = 0;
    if (expert_mode) { while (e < N_EXPERTS - 1 && rowbase >= meta[M_OFF + e + 1]) ++e; }
    const size_t wb = (size_t)e * D_MODEL * D_FF;

    const int lr = lane & 15, hi = lane >> 4;
    const int colbase = blockIdx.y * 256 + wave * 64;
    const unsigned short* hp = H  + (size_t)(rowbase + lr) * D_FF;
    const unsigned short* wp = Wd + wb + (size_t)(colbase + lr) * D_FF;

    v8f acc[4] = {};
    for (int k0 = 0; k0 < D_FF; k0 += 32) {
        const int c0 = k0 + hi * 8, c1 = c0 + 16;
        Frag16 a, b[4];
        a.q[0] = *(const u32x4*)(hp + c0);
        a.q[1] = *(const u32x4*)(hp + c1);
#pragma unroll
        for (int j = 0; j < 4; ++j) {
            const size_t joff = (size_t)j * 16 * D_FF;
            b[j].q[0] = *(const u32x4*)(wp + joff + c0);
            b[j].q[1] = *(const u32x4*)(wp + joff + c1);
        }
#pragma unroll
        for (int j = 0; j < 4; ++j) {
            acc[j] = __builtin_amdgcn_wmma_f32_16x16x32_bf16(false, a.v, false, b[j].v,
                                                             (short)0, acc[j], false, false);
        }
    }
#pragma unroll
    for (int j = 0; j < 4; ++j) {
        const int colo = colbase + j * 16 + lr;
#pragma unroll
        for (int i = 0; i < 8; ++i) {
            const int row = rowbase + i + hi * 8;
            float val = acc[j][i];
            if (expert_mode) {
                const int tok = row_token[row];
                if (tok >= 0)
                    atomicAdd(out + (size_t)tok * D_MODEL + colo, val * wrow[row]);
            } else {
                atomicAdd(out + (size_t)row * D_MODEL + colo, val);
            }
        }
    }
}

// ---------------- host launcher ----------------
extern "C" void kernel_launch(void* const* d_in, const int* in_sizes, int n_in,
                              void* d_out, int out_size, void* d_ws, size_t ws_size,
                              hipStream_t stream) {
    const float* x        = (const float*)d_in[0];
    const float* router_w = (const float*)d_in[1];
    const float* w_gate   = (const float*)d_in[2];
    const float* w_up     = (const float*)d_in[3];
    const float* w_down   = (const float*)d_in[4];
    const float* sg       = (const float*)d_in[5];
    const float* su       = (const float*)d_in[6];
    const float* sd       = (const float*)d_in[7];
    float* out = (float*)d_out;

    char* ws = (char*)d_ws;
    size_t o = 0;
    auto take = [&](size_t bytes) { size_t p = o; o += (bytes + 255) & ~(size_t)255; return p; };

    int*            meta      = (int*)           (ws + take(256));
    int*            tok_e     = (int*)           (ws + take((size_t)2 * N_TOK * 4));
    float*          tok_w     = (float*)         (ws + take((size_t)2 * N_TOK * 4));
    int*            row_token = (int*)           (ws + take((size_t)MAX_ROWS * 4));
    float*          wrow      = (float*)         (ws + take((size_t)MAX_ROWS * 4));
    unsigned short* xg        = (unsigned short*)(ws + take((size_t)MAX_ROWS * D_MODEL * 2));
    unsigned short* xb        = (unsigned short*)(ws + take((size_t)N_TOK * D_MODEL * 2));
    unsigned short* hg        = (unsigned short*)(ws + take((size_t)MAX_ROWS * D_FF * 2));
    unsigned short* hs        = (unsigned short*)(ws + take((size_t)N_TOK * D_FF * 2));
    unsigned short* wgT       = (unsigned short*)(ws + take((size_t)N_EXPERTS * D_MODEL * D_FF * 2));
    unsigned short* wuT       = (unsigned short*)(ws + take((size_t)N_EXPERTS * D_MODEL * D_FF * 2));
    unsigned short* wdT       = (unsigned short*)(ws + take((size_t)N_EXPERTS * D_MODEL * D_FF * 2));
    unsigned short* sgT       = (unsigned short*)(ws + take((size_t)D_MODEL * D_FF * 2));
    unsigned short* suT       = (unsigned short*)(ws + take((size_t)D_MODEL * D_FF * 2));
    unsigned short* sdT       = (unsigned short*)(ws + take((size_t)D_MODEL * D_FF * 2));

    hipMemsetAsync(meta, 0, 256, stream);
    hipMemsetAsync(row_token, 0xFF, (size_t)MAX_ROWS * 4, stream);            // token = -1
    hipMemsetAsync(xg, 0, (size_t)MAX_ROWS * D_MODEL * 2, stream);            // zero pad rows
    hipMemsetAsync(out, 0, (size_t)out_size * sizeof(float), stream);

    // routing
    k_router<<<N_TOK / 8, 256, 0, stream>>>(x, router_w, meta, tok_e, tok_w);
    k_offsets<<<1, 1, 0, stream>>>(meta);
    k_gather<<<N_TOK, 64, 0, stream>>>(x, tok_e, tok_w, meta, row_token, wrow, xg, xb);

    // weight conversion (one pass; ~113 MB bf16 result fits the 192 MB L2)
    const int gDF = (D_MODEL * D_FF + 255) / 256;
    k_tcvt<<<dim3(gDF, 1, N_EXPERTS), 256, 0, stream>>>(w_gate, wgT, D_MODEL, D_FF);
    k_tcvt<<<dim3(gDF, 1, N_EXPERTS), 256, 0, stream>>>(w_up,   wuT, D_MODEL, D_FF);
    k_tcvt<<<dim3(gDF, 1, N_EXPERTS), 256, 0, stream>>>(w_down, wdT, D_FF,    D_MODEL);
    k_tcvt<<<dim3(gDF, 1, 1),         256, 0, stream>>>(sg,     sgT, D_MODEL, D_FF);
    k_tcvt<<<dim3(gDF, 1, 1),         256, 0, stream>>>(su,     suT, D_MODEL, D_FF);
    k_tcvt<<<dim3(gDF, 1, 1),         256, 0, stream>>>(sd,     sdT, D_FF,    D_MODEL);

    // routed experts: worst-case fixed grid, device-side early exit past meta[M_TOT]
    k_gateup<<<dim3(ROW_TILES, D_FF / 256), 128, 0, stream>>>(xg, wgT, wuT, hg, meta, 1, 0);
    k_down  <<<dim3(ROW_TILES, D_MODEL / 256), 128, 0, stream>>>(hg, wdT, out, meta, row_token, wrow, 1, 0);

    // shared expert over all tokens
    k_gateup<<<dim3(N_TOK / 16, D_FF / 256), 128, 0, stream>>>(xb, sgT, suT, hs, meta, 0, N_TOK);
    k_down  <<<dim3(N_TOK / 16, D_MODEL / 256), 128, 0, stream>>>(hs, sdT, out, meta, row_token, wrow, 0, N_TOK);
}
